// RoPEAttention_52347061403920
// MI455X (gfx1250) — matrix operand order repaired and checked by reading, not compiled, not run on hardware
//
#include <hip/hip_runtime.h>
#include <hip/hip_bf16.h>

#define B_SZ   2
#define S_LEN  2048
#define F_DIM  2048
#define NH     16
#define DHEAD  128
#define NCOL   2048   // NH*DHEAD

typedef __attribute__((ext_vector_type(16))) __bf16          bf16x16;
typedef __attribute__((ext_vector_type(8)))  __bf16          bf16x8;
typedef __attribute__((ext_vector_type(8)))  float           f32x8;
typedef __attribute__((ext_vector_type(4)))  float           f32x4;
typedef __attribute__((ext_vector_type(8)))  unsigned short  u16x8;
typedef __attribute__((ext_vector_type(4)))  unsigned int    u32x4;
typedef __attribute__((ext_vector_type(8)))  int             i32x8;
typedef __attribute__((ext_vector_type(4)))  int             i32x4;

#define F32X8_ZERO ((f32x8){0.f,0.f,0.f,0.f,0.f,0.f,0.f,0.f})

__device__ __forceinline__ unsigned short f2bf(float f) {
  unsigned u = __float_as_uint(f);
  u += 0x7FFFu + ((u >> 16) & 1u);          // round-to-nearest-even
  return (unsigned short)(u >> 16);
}

// Build a v16bf A/B fragment from two 16-byte chunks (byte offsets off0/off1).
__device__ __forceinline__ bf16x16 ld_frag(const unsigned short* p, int off0, int off1) {
  const char* cp = (const char*)p;
  union { bf16x16 v; bf16x8 h[2]; } u;
  u.h[0] = *(const bf16x8*)(cp + off0);
  u.h[1] = *(const bf16x8*)(cp + off1);
  return u.v;
}

__device__ __forceinline__ f32x8 wmma_bf16(bf16x16 a, bf16x16 b, f32x8 c) {
  return __builtin_amdgcn_wmma_f32_16x16x32_bf16(false, a, false, b, (short)0, c,
                                                 false, false);
}

// ---------------------------------------------------------------------------
// Kernel 1: fused QKV projection + bias + RoPE, fp32 in -> bf16 out.
// grid = (NCOL/128, (B*S)/128, 3), block = 256 (8 wave32).
// ---------------------------------------------------------------------------
__global__ __launch_bounds__(256) void qkv_rope_kernel(
    const float* __restrict__ X,
    const float* __restrict__ Wq, const float* __restrict__ Wk,
    const float* __restrict__ Wv,
    const float* __restrict__ bq, const float* __restrict__ bk,
    const float* __restrict__ bv,
    unsigned short* __restrict__ Qo, unsigned short* __restrict__ Ko,
    unsigned short* __restrict__ Vo,
    const int* __restrict__ offset_p)
{
  // Atile [128][56] u16 (stride 112B) and Wt [128][56] u16 overlap the
  // f32 epilogue tile Ct [128][132] (stride 528B).
  __shared__ __align__(16) unsigned char lds[128 * 132 * 4];
  unsigned short* At = (unsigned short*)lds;                 // rows = M, cols = k
  unsigned short* Wt = (unsigned short*)(lds + 128 * 56 * 2);// rows = N, cols = k
  float* Ct = (float*)lds;                                   // [128][132]

  const int z = blockIdx.z;
  const float* W    = (z == 0) ? Wq : (z == 1) ? Wk : Wv;
  const float* bias = (z == 0) ? bq : (z == 1) ? bk : bv;
  unsigned short* Out = (z == 0) ? Qo : (z == 1) ? Ko : Vo;

  const int n0 = blockIdx.x * 128;       // one full head (RoPE pair in-tile)
  const int m0 = blockIdx.y * 128;
  const int t = threadIdx.x;
  const int w = t >> 5, lane = t & 31, lo = lane & 15, hi = lane >> 4;
  const int wm = w >> 1, wn = w & 1;     // wave tile: M 32 x N 64

  f32x8 acc[2][4];
#pragma unroll
  for (int i = 0; i < 2; ++i)
#pragma unroll
    for (int j = 0; j < 4; ++j) acc[i][j] = F32X8_ZERO;

  for (int k0 = 0; k0 < F_DIM; k0 += 32) {
    // ---- stage A tile 128x32 f32 -> bf16 LDS -----------------------------
#pragma unroll
    for (int p = 0; p < 4; ++p) {
      int row = p * 32 + (t >> 3);
      int col = (t & 7) * 4;
      f32x4 v = *(const f32x4*)(X + (size_t)(m0 + row) * F_DIM + k0 + col);
      unsigned long long pk =
          (unsigned long long)f2bf(v.x)          |
          ((unsigned long long)f2bf(v.y) << 16)  |
          ((unsigned long long)f2bf(v.z) << 32)  |
          ((unsigned long long)f2bf(v.w) << 48);
      *(unsigned long long*)(At + row * 56 + col) = pk;
    }
    // ---- stage W tile 32x128 f32 -> transposed bf16 LDS ------------------
#pragma unroll
    for (int p = 0; p < 4; ++p) {
      int kr  = p * 8 + (t >> 5);
      int col = (t & 31) * 4;
      f32x4 v = *(const f32x4*)(W + (size_t)(k0 + kr) * NCOL + n0 + col);
      Wt[(col + 0) * 56 + kr] = f2bf(v.x);
      Wt[(col + 1) * 56 + kr] = f2bf(v.y);
      Wt[(col + 2) * 56 + kr] = f2bf(v.z);
      Wt[(col + 3) * 56 + kr] = f2bf(v.w);
    }
    // ---- prefetch next k-step tiles (L2-resident; emits global_prefetch) --
    if (k0 + 32 < F_DIM) {
      __builtin_prefetch(X + (size_t)(m0 + (t >> 3)) * F_DIM + k0 + 32 + (t & 7) * 4, 0, 1);
      __builtin_prefetch(W + (size_t)(k0 + 32 + (t >> 5)) * NCOL + n0 + (t & 31) * 4, 0, 1);
    }
    __syncthreads();

    // ---- 8 WMMAs per wave per k-step -------------------------------------
    const unsigned short* ar0 = At + (wm * 32 + lo) * 56;
    const unsigned short* ar1 = At + (wm * 32 + 16 + lo) * 56;
    bf16x16 a0 = ld_frag(ar0, hi * 16, hi * 16 + 32);   // A layout: K {0-7,16-23}/{8-15,24-31}
    bf16x16 a1 = ld_frag(ar1, hi * 16, hi * 16 + 32);
#pragma unroll
    for (int nt = 0; nt < 4; ++nt) {
      const unsigned short* br = Wt + (wn * 64 + nt * 16 + lo) * 56;
      bf16x16 bfr = ld_frag(br, hi * 16, hi * 16 + 32); // B layout: K {0-7,16-23}/{8-15,24-31}
      acc[0][nt] = wmma_bf16(a0, bfr, acc[0][nt]);
      acc[1][nt] = wmma_bf16(a1, bfr, acc[1][nt]);
    }
    __syncthreads();
  }

  // ---- dump accumulators to f32 LDS tile (C layout: M = r + 8*hi, N = lo) --
#pragma unroll
  for (int mi = 0; mi < 2; ++mi)
#pragma unroll
    for (int nt = 0; nt < 4; ++nt)
#pragma unroll
      for (int r = 0; r < 8; ++r) {
        int row = wm * 32 + mi * 16 + r + hi * 8;
        int col = wn * 64 + nt * 16 + lo;
        Ct[row * 132 + col] = acc[mi][nt][r];
      }
  __syncthreads();

  // ---- bias + RoPE epilogue, write bf16 ------------------------------------
  const int off = *offset_p;
  const bool do_rope = (z < 2);
  for (int i = t; i < 128 * 64; i += 256) {
    int row = i >> 6;
    int d   = i & 63;                 // frequency index j = d
    int m   = m0 + row;
    int s   = m & (S_LEN - 1);        // s within batch
    float c1 = Ct[row * 132 + d]      + bias[n0 + d];
    float c2 = Ct[row * 132 + d + 64] + bias[n0 + d + 64];
    unsigned short e1, e2;
    if (do_rope) {
      // inv_freq = 10000^(-d/64) = exp(-ln(10000)/64 * d)
      float ang = (float)(s + off) * __expf(-0.14391156514261f * (float)d);
      float sn, cs;
      __sincosf(ang, &sn, &cs);
      e1 = f2bf(c1 * cs - c2 * sn);
      e2 = f2bf(c1 * sn + c2 * cs);
    } else {
      e1 = f2bf(c1);
      e2 = f2bf(c2);
    }
    Out[(size_t)m * NCOL + n0 + d]      = e1;
    Out[(size_t)m * NCOL + n0 + d + 64] = e2;
  }
}

// ---------------------------------------------------------------------------
// Kernel 2: causal flash attention, bf16 WMMA, f32 online softmax.
// K tile staged by the Tensor Data Mover (TDM); V tile staged+transposed by
// all waves; wave 0 joins after its tensor wait.
// grid = (S/128, NH, B), block = 256 (8 wave32, 16 q-rows per wave).
// ---------------------------------------------------------------------------
__global__ __launch_bounds__(256) void fattn_kernel(
    const unsigned short* __restrict__ Q,
    const unsigned short* __restrict__ K,
    const unsigned short* __restrict__ V,
    float* __restrict__ O)
{
  __shared__ __align__(16) unsigned char lds[64 * 136 * 2 + 2 * 128 * 72 * 2];
  unsigned short* Kt = (unsigned short*)lds;                       // [64][136] row-major
  unsigned short* Vt = (unsigned short*)(lds + 64 * 136 * 2);      // [128][72] transposed (row=d)
  unsigned short* Pt = (unsigned short*)(lds + 64 * 136 * 2 + 128 * 72 * 2); // [128][72]

  const int q0 = blockIdx.x * 128;
  const int h  = blockIdx.y;
  const int b  = blockIdx.z;
  const int t = threadIdx.x;
  const int w = t >> 5, lane = t & 31, lo = lane & 15, hi = lane >> 4;

  const size_t head_off = ((size_t)b * S_LEN * NH + h) * DHEAD;
  const unsigned short* Qb = Q + head_off;   // row stride NCOL
  const unsigned short* Kb = K + head_off;
  const unsigned short* Vb = V + head_off;

  // Q A-fragments for this wave's 16 rows (loaded once, kept in VGPRs).
  bf16x16 qf[4];
  {
    const unsigned short* qp = Qb + (size_t)(q0 + w * 16 + lo) * NCOL;
#pragma unroll
    for (int ks = 0; ks < 4; ++ks)
      qf[ks] = ld_frag(qp, ks * 64 + hi * 16, ks * 64 + hi * 16 + 32);
  }

  // Constant half of the TDM descriptor (group1: sizes/strides/padding).
  // data_size=2B (code 1); pad_enable; pad_interval = 64 DWORDs (code 5, i.e.
  // every 256B = one 128-elem row); pad_amount = 4 DWORDs (code 3, 16B) ->
  // reproduces the 136-u16 LDS row stride of Kt.
  i32x8 tdm_g1;
  tdm_g1[0] = (1 << 16) | (1 << 20) | (5 << 22) | (3 << 25);
  tdm_g1[1] = (int)((unsigned)DHEAD << 16);      // tensor_dim0[15:0] @ bits 63:48
  tdm_g1[2] = (int)((unsigned)S_LEN << 16);      // tensor_dim1[15:0] @ bits 95:80
  tdm_g1[3] = (int)((unsigned)DHEAD << 16);      // tile_dim0 @ bits 127:112
  tdm_g1[4] = 64;                                // tile_dim1 = 64 rows, tile_dim2 = 0
  tdm_g1[5] = NCOL;                              // tensor_dim0_stride[31:0] = 2048
  tdm_g1[6] = 0;                                 // stride hi / dim1_stride lo
  tdm_g1[7] = 0;
  const i32x4 tdm_zero4 = (i32x4){0, 0, 0, 0};   // groups 2/3: 2-D tensor
  const i32x8 tdm_zero8 = (i32x8){0, 0, 0, 0, 0, 0, 0, 0};
  const unsigned kt_lds = (unsigned)(unsigned long long)(void*)Kt;

  float mrow[8], lrow[8];
  f32x8 oacc[8];
#pragma unroll
  for (int r = 0; r < 8; ++r) { mrow[r] = -1e30f; lrow[r] = 0.f; }
#pragma unroll
  for (int dt = 0; dt < 8; ++dt) oacc[dt] = F32X8_ZERO;

  const float scale = 0.08838834764831845f;   // 1/sqrt(128)
  const int kv_end = q0 + 128;                // causal: skip tiles above diagonal

  for (int kv0 = 0; kv0 < kv_end; kv0 += 64) {
    __syncthreads();                          // WAR vs previous iteration's reads
    // ---- K tile: 64x128 bf16 via TENSOR_LOAD_TO_LDS (wave 0 only) --------
    if (w == 0) {
      unsigned long long ga =
          (unsigned long long)(const void*)(Kb + (size_t)kv0 * NCOL);
      u32x4 g0;
      g0[0] = 1u;                                              // count=1
      g0[1] = kt_lds;                                          // lds_addr
      g0[2] = (unsigned)(ga & 0xffffffffu);                    // global_addr lo
      g0[3] = (unsigned)((ga >> 32) & 0x1ffffffu) | (2u << 30);// addr hi | type=2
      __builtin_amdgcn_tensor_load_to_lds(g0, tdm_g1, tdm_zero4, tdm_zero4,
                                          tdm_zero8, 0);
      __builtin_amdgcn_s_wait_tensorcnt(0);
    }
    // ---- V tile: 64x128 -> transposed [d][kv] (all waves) ----------------
#pragma unroll
    for (int p = 0; p < 4; ++p) {
      int c = p * 256 + t;
      int row = c >> 4, ch = c & 15;
      u16x8 v = *(const u16x8*)(Vb + (size_t)(kv0 + row) * NCOL + ch * 8);
#pragma unroll
      for (int i = 0; i < 8; ++i)
        Vt[(ch * 8 + i) * 72 + row] = v[i];
    }
    __syncthreads();   // publishes TDM completion + V transpose to all waves

    // ---- S = Q K^T : 16 WMMAs --------------------------------------------
    f32x8 sacc[4];
#pragma unroll
    for (int nt = 0; nt < 4; ++nt) sacc[nt] = F32X8_ZERO;
#pragma unroll
    for (int ks = 0; ks < 4; ++ks)
#pragma unroll
      for (int nt = 0; nt < 4; ++nt) {
        const unsigned short* kr = Kt + (nt * 16 + lo) * 136;  // column N = kv row
        bf16x16 kf = ld_frag(kr, ks * 64 + hi * 16, ks * 64 + hi * 16 + 32);
        sacc[nt] = wmma_bf16(qf[ks], kf, sacc[nt]);
      }

    // ---- online softmax (row stats via 16-lane xor reductions) ------------
#pragma unroll
    for (int r = 0; r < 8; ++r) {
      int qi = q0 + w * 16 + r + 8 * hi;
      float vmax = -1e30f;
#pragma unroll
      for (int nt = 0; nt < 4; ++nt) {
        float sv = sacc[nt][r] * scale;
        int kvi = kv0 + nt * 16 + lo;
        sv = (kvi > qi) ? -1e30f : sv;        // causal mask
        sacc[nt][r] = sv;
        vmax = fmaxf(vmax, sv);
      }
#pragma unroll
      for (int m = 8; m >= 1; m >>= 1)
        vmax = fmaxf(vmax, __shfl_xor(vmax, m, 32));
      float mnew  = fmaxf(mrow[r], vmax);
      float alpha = __expf(mrow[r] - mnew);
      float rs = 0.f;
#pragma unroll
      for (int nt = 0; nt < 4; ++nt) {
        float pv = __expf(sacc[nt][r] - mnew);
        sacc[nt][r] = pv;
        rs += pv;
      }
#pragma unroll
      for (int m = 8; m >= 1; m >>= 1)
        rs += __shfl_xor(rs, m, 32);
      lrow[r] = lrow[r] * alpha + rs;
      mrow[r] = mnew;
#pragma unroll
      for (int dt = 0; dt < 8; ++dt)
        oacc[dt][r] = oacc[dt][r] * alpha;
    }

    // ---- P -> LDS (bf16) --------------------------------------------------
#pragma unroll
    for (int nt = 0; nt < 4; ++nt)
#pragma unroll
      for (int r = 0; r < 8; ++r)
        Pt[(w * 16 + r + 8 * hi) * 72 + nt * 16 + lo] = f2bf(sacc[nt][r]);
    __syncthreads();

    // ---- O += P V : 16 WMMAs ---------------------------------------------
    const unsigned short* prow = Pt + (w * 16 + lo) * 72;
#pragma unroll
    for (int ks = 0; ks < 2; ++ks) {
      bf16x16 pf = ld_frag(prow, ks * 64 + hi * 16, ks * 64 + hi * 16 + 32);
#pragma unroll
      for (int dt = 0; dt < 8; ++dt) {
        const unsigned short* vr = Vt + (dt * 16 + lo) * 72; // column N = d
        bf16x16 vf = ld_frag(vr, ks * 64 + hi * 16, ks * 64 + hi * 16 + 32);
        oacc[dt] = wmma_bf16(pf, vf, oacc[dt]);
      }
    }
  }

  // ---- normalize and write f32 output --------------------------------------
#pragma unroll
  for (int r = 0; r < 8; ++r) {
    int qi = q0 + w * 16 + r + 8 * hi;
    float inv = 1.f / lrow[r];
    float* op = O + ((size_t)(b * S_LEN + qi)) * NCOL + h * DHEAD;
#pragma unroll
    for (int dt = 0; dt < 8; ++dt)
      op[dt * 16 + lo] = oacc[dt][r] * inv;
  }
}

// ---------------------------------------------------------------------------
// Host launcher.
// Inputs (setup_inputs order): in_q, Wq, bq, Wk, bk, Wv, bv, mask, offset
// ---------------------------------------------------------------------------
extern "C" void kernel_launch(void* const* d_in, const int* in_sizes, int n_in,
                              void* d_out, int out_size, void* d_ws, size_t ws_size,
                              hipStream_t stream) {
  (void)in_sizes; (void)n_in; (void)out_size; (void)ws_size;

  const float* X  = (const float*)d_in[0];
  const float* Wq = (const float*)d_in[1];
  const float* bq = (const float*)d_in[2];
  const float* Wk = (const float*)d_in[3];
  const float* bk = (const float*)d_in[4];
  const float* Wv = (const float*)d_in[5];
  const float* bv = (const float*)d_in[6];
  // d_in[7] = mask (causality computed analytically), d_in[8] = offset
  const int* offset = (const int*)d_in[8];

  const size_t elems = (size_t)B_SZ * S_LEN * NH * DHEAD;   // 8Mi elements each
  unsigned short* qb = (unsigned short*)d_ws;
  unsigned short* kb = qb + elems;
  unsigned short* vb = kb + elems;

  dim3 g1(NCOL / 128, (B_SZ * S_LEN) / 128, 3);
  qkv_rope_kernel<<<g1, 256, 0, stream>>>(X, Wq, Wk, Wv, bq, bk, bv,
                                          qb, kb, vb, offset);

  dim3 g2(S_LEN / 128, NH, B_SZ);
  fattn_kernel<<<g2, 256, 0, stream>>>(qb, kb, vb, (float*)d_out);
}
